// DSVT_ort_59304908423227
// MI455X (gfx1250) — matrix operand order repaired and checked by reading, not compiled, not run on hardware
//
#include <hip/hip_runtime.h>
#include <math.h>

// ---------------- problem constants ----------------
#define NVOX 36864
#define CDIM 256
#define FDIM 512
#define KSET 36
#define SSET 1024
#define NHEADS 8
#define DH 32
#define LLAY 4

typedef __attribute__((ext_vector_type(16))) __bf16 v16bf;
typedef __attribute__((ext_vector_type(8)))  float  v8f;
typedef __attribute__((ext_vector_type(4)))  __bf16 bf4;

static __device__ __forceinline__ v8f wmma_bf16(v16bf a, v16bf b, v8f c) {
  // D = A(16x32 bf16) * B(32x16 bf16) + C(f32), f32 accumulate
  return __builtin_amdgcn_wmma_f32_16x16x32_bf16(false, a, false, b, (short)0, c,
                                                 false, false);
}

static __device__ __forceinline__ v8f zero_v8() {
  v8f z;
#pragma unroll
  for (int i = 0; i < 8; ++i) z[i] = 0.f;
  return z;
}
static __device__ __forceinline__ v16bf zero_v16() {
  v16bf z;
#pragma unroll
  for (int i = 0; i < 16; ++i) z[i] = (__bf16)0.f;
  return z;
}

// A-matrix (16x32, MxK) lane/VGPR K mapping (matches B's contiguous order):
// lanes 0-15: M=lane, K 0..15; lanes 16-31: M=lane-16, K 16..31
static __device__ __forceinline__ int amap(int e, int half) {
  return 2 * e + 16 * half;
}

static __device__ __forceinline__ float gelu_f(float v) {
  return 0.5f * v * (1.f + erff(v * 0.7071067811865475f));
}

static __device__ __forceinline__ float wave_sum(float v) {
#pragma unroll
  for (int o = 16; o > 0; o >>= 1) v += __shfl_xor(v, o, 32);
  return v;
}

// low 32 bits of a generic pointer to __shared__ == LDS byte offset
static __device__ __forceinline__ unsigned lds_addr(const void* p) {
  return (unsigned)(unsigned long long)p;
}

// GLOBAL_LOAD_ASYNC_TO_LDS_B128: DMA 16B/lane global -> LDS, tracked by ASYNCcnt
static __device__ __forceinline__ void async_b128(unsigned lds, const void* g) {
  asm volatile("global_load_async_to_lds_b128 %0, %1, off"
               :: "v"(lds), "v"(g)
               : "memory");
}
static __device__ __forceinline__ void wait_async0() {
  asm volatile("s_wait_asynccnt 0" ::: "memory");
}

// ---------------- generic bf16 GEMM: Y = A(MxK) * W(NoutxK)^T + bias ----------------
// block tile 256(M) x 64(N) x 32(K); 8 waves, each 64x32 (4x2 WMMA tiles);
// double-buffered LDS staged with async-to-LDS DMA, one barrier per K-step.
#define BM 256
#define BN 64
#define BK 32
#define LDT 40  // padded LDS row stride (bf16 elems) -> 80B rows, 16B-aligned chunks

#define GF_OUT_BF16 1
#define GF_GELU 2

static __device__ __forceinline__ v16bf lds_load_a(const __bf16* rowbase, int lane) {
  const int m = lane & 15, half = lane >> 4;
  const __bf16* r = rowbase + m * LDT;
  v16bf a;
#pragma unroll
  for (int e = 0; e < 8; ++e) {
    const int kb = amap(e, half);
    a[2 * e] = r[kb];
    a[2 * e + 1] = r[kb + 1];
  }
  return a;
}
// B-matrix (32x16, KxN): lane&15 = column; lanes 0-15 K=0..15, lanes 16-31 K=16..31
static __device__ __forceinline__ v16bf lds_load_b(const __bf16* rowbase, int lane) {
  const int n = lane & 15, kh = lane >> 4;
  const __bf16* r = rowbase + n * LDT;  // LDS holds W rows (col-of-B == row-of-W)
  v16bf b;
#pragma unroll
  for (int e = 0; e < 8; ++e) {
    const int kb = 16 * kh + 2 * e;
    b[2 * e] = r[kb];
    b[2 * e + 1] = r[kb + 1];
  }
  return b;
}

__global__ __launch_bounds__(256) void gemm_bf16_kernel(
    const __bf16* __restrict__ A, const __bf16* __restrict__ W,
    const float* __restrict__ bias, void* __restrict__ Y,
    const int* __restrict__ scat, int Kdim, int ldY, int flags) {
  __shared__ alignas(16) __bf16 As[2][BM * LDT];  // 2 x 20 KB
  __shared__ alignas(16) __bf16 Bs[2][BN * LDT];  // 2 x  5 KB
  const int tid = threadIdx.x, lane = tid & 31, wid = tid >> 5;
  const int wm = (wid & 3) * 64, wn = (wid >> 2) * 32;  // 4x2 wave grid
  const long mBase = (long)blockIdx.y * BM;
  const long nBase = (long)blockIdx.x * BN;

  v8f acc[4][2];
#pragma unroll
  for (int mi = 0; mi < 4; ++mi)
#pragma unroll
    for (int ni = 0; ni < 2; ++ni) acc[mi][ni] = zero_v8();

  // DMA mapping: A -> one 32-elem row per thread (4 x b128); B -> 8 elems/thread
  const int brow = tid >> 2, bcol = (tid & 3) * 8;
  const __bf16* gA = A + (mBase + tid) * (long)Kdim;
  const __bf16* gB = W + (nBase + brow) * (long)Kdim + bcol;
  unsigned laBase[2], lbBase[2];
#pragma unroll
  for (int p = 0; p < 2; ++p) {
    laBase[p] = lds_addr(&As[p][0]) + (unsigned)tid * (LDT * 2);
    lbBase[p] = lds_addr(&Bs[p][0]) + (unsigned)brow * (LDT * 2) + (unsigned)bcol * 2;
  }

  // prologue: DMA first K-slab into buffer 0
  async_b128(laBase[0] + 0,  gA + 0);
  async_b128(laBase[0] + 16, gA + 8);
  async_b128(laBase[0] + 32, gA + 16);
  async_b128(laBase[0] + 48, gA + 24);
  async_b128(lbBase[0],      gB);
  wait_async0();
  __syncthreads();

  int p = 0;
  for (int k0 = 0; k0 < Kdim; k0 += BK) {
    // kick off DMA of next K-slab into the other buffer (overlaps the WMMAs)
    if (k0 + BK < Kdim) {
      const __bf16* nA = gA + k0 + BK;
      const __bf16* nB = gB + k0 + BK;
      const int q = p ^ 1;
      async_b128(laBase[q] + 0,  nA + 0);
      async_b128(laBase[q] + 16, nA + 8);
      async_b128(laBase[q] + 32, nA + 16);
      async_b128(laBase[q] + 48, nA + 24);
      async_b128(lbBase[q],      nB);
    }
    // compute on current buffer
    v16bf fb0 = lds_load_b(&Bs[p][(wn + 0) * LDT], lane);
    v16bf fb1 = lds_load_b(&Bs[p][(wn + 16) * LDT], lane);
#pragma unroll
    for (int mi = 0; mi < 4; ++mi) {
      const v16bf fa = lds_load_a(&As[p][(wm + mi * 16) * LDT], lane);
      acc[mi][0] = wmma_bf16(fa, fb0, acc[mi][0]);
      acc[mi][1] = wmma_bf16(fa, fb1, acc[mi][1]);
    }
    wait_async0();     // next slab fully landed in LDS
    __syncthreads();   // all waves done reading buf[p]; publish buf[p^1]
    p ^= 1;
  }

  // epilogue: C/D layout — col = lane&15, row = e + 8*(lane>>4)
  const int nit = lane & 15, rh = lane >> 4;
#pragma unroll
  for (int mi = 0; mi < 4; ++mi)
#pragma unroll
    for (int ni = 0; ni < 2; ++ni) {
      const v8f c = acc[mi][ni];
#pragma unroll
      for (int e = 0; e < 8; ++e) {
        const long m = mBase + wm + mi * 16 + e + 8 * rh;
        const long n = nBase + wn + ni * 16 + nit;
        float v = c[e] + bias[n];
        if (flags & GF_GELU) v = gelu_f(v);
        const long row = scat ? (long)scat[m] : m;
        if (flags & GF_OUT_BF16)
          ((__bf16*)Y)[row * (long)ldY + n] = (__bf16)v;
        else
          ((float*)Y)[row * (long)ldY + n] = v;
      }
    }
}

// ---------------- attention: 1 wave = 1 (set, head) ----------------
#define PP 52  // padded f32 LDS stride for 48x48 score tile

__global__ __launch_bounds__(128) void attn_kernel(
    const __bf16* __restrict__ qkv,            // (N, 768) : q|k|v
    const unsigned char* __restrict__ msk,     // (S, 36) key mask
    __bf16* __restrict__ outb) {               // (N, 256)
  __shared__ float Pf[4][48 * PP];
  __shared__ float rsum[4][48];
  const int lane = threadIdx.x & 31;
  const int wv = threadIdx.x >> 5;
  const int head = blockIdx.y * 4 + wv;
  const int s = blockIdx.x;
  const __bf16* base = qkv + (size_t)s * KSET * 768;
  const unsigned char* km = msk + (size_t)s * KSET;
  float* P = Pf[wv];
  float* rs = rsum[wv];
  const int nit = lane & 15, half = lane >> 4;

  // ---- scores = Q @ K^T, tokens padded 36->48, Dh = 32 (one WMMA K-step) ----
  v16bf aQ[3], bK[3];
#pragma unroll
  for (int t = 0; t < 3; ++t) {
    const int tok = t * 16 + nit;
    v16bf a = zero_v16(), b = zero_v16();
    if (tok < KSET) {
      const __bf16* rq = base + (size_t)tok * 768 + head * DH;        // q cols
      const __bf16* rk = base + (size_t)tok * 768 + 256 + head * DH;  // k cols
#pragma unroll
      for (int e = 0; e < 8; ++e) {
        const int ka = amap(e, half);
        a[2 * e] = rq[ka];
        a[2 * e + 1] = rq[ka + 1];
        const int kb = 16 * half + 2 * e;
        b[2 * e] = rk[kb];
        b[2 * e + 1] = rk[kb + 1];
      }
    }
    aQ[t] = a;
    bK[t] = b;
  }
  v8f sc[3][3];
#pragma unroll
  for (int tm = 0; tm < 3; ++tm)
#pragma unroll
    for (int tn = 0; tn < 3; ++tn) {
      sc[tm][tn] = zero_v8();
      sc[tm][tn] = wmma_bf16(aQ[tm], bK[tn], sc[tm][tn]);
    }

  // ---- masked scores -> LDS ----
  const float scale = 0.17677669529663687f;  // 32^-0.5
  unsigned char mloc[3];
  mloc[0] = km[nit];
  mloc[1] = km[nit + 16];
  mloc[2] = (nit + 32 < KSET) ? km[nit + 32] : (unsigned char)1;  // pad keys masked
#pragma unroll
  for (int tm = 0; tm < 3; ++tm)
#pragma unroll
    for (int tn = 0; tn < 3; ++tn)
#pragma unroll
      for (int e = 0; e < 8; ++e) {
        const int m = tm * 16 + e + 8 * half;
        const int n = tn * 16 + nit;
        P[m * PP + n] = mloc[tn] ? -1e9f : sc[tm][tn][e] * scale;
      }
  asm volatile("s_wait_dscnt 0" ::: "memory");

  // ---- softmax rows (each lane: rows lane, lane+32) ----
  for (int m = lane; m < 48; m += 32) {
    if (m < KSET) {
      float mx = -1e30f;
      for (int n = 0; n < 48; ++n) mx = fmaxf(mx, P[m * PP + n]);
      float sum = 0.f;
      for (int n = 0; n < 48; ++n) {
        const float e = __expf(P[m * PP + n] - mx);
        P[m * PP + n] = e;
        sum += e;
      }
      rs[m] = 1.f / sum;
    } else {
      for (int n = 0; n < 48; ++n) P[m * PP + n] = 0.f;
      rs[m] = 0.f;
    }
  }
  asm volatile("s_wait_dscnt 0" ::: "memory");

  // ---- out = P(48x64, zero-padded) @ V(64x32) ----
  v8f o[3][2];
#pragma unroll
  for (int tm = 0; tm < 3; ++tm)
#pragma unroll
    for (int tn = 0; tn < 2; ++tn) o[tm][tn] = zero_v8();
  const __bf16* vb = base + 512 + head * DH;
  for (int ko = 0; ko < 64; ko += 32) {
    v16bf aP[3];
#pragma unroll
    for (int t = 0; t < 3; ++t) {
      const int m = t * 16 + nit;
      const float r = rs[m];
      v16bf a;
#pragma unroll
      for (int e = 0; e < 8; ++e) {
        const int kb = amap(e, half) + ko;
        const float v0 = (kb < 48) ? P[m * PP + kb] * r : 0.f;
        const float v1 = (kb + 1 < 48) ? P[m * PP + kb + 1] * r : 0.f;
        a[2 * e] = (__bf16)v0;
        a[2 * e + 1] = (__bf16)v1;
      }
      aP[t] = a;
    }
    v16bf bV[2];
#pragma unroll
    for (int t = 0; t < 2; ++t) {
      v16bf b;
#pragma unroll
      for (int e = 0; e < 8; ++e) {
        const int t0 = ko + 16 * half + 2 * e;
        b[2 * e] = (t0 < KSET) ? vb[(size_t)t0 * 768 + t * 16 + nit] : (__bf16)0.f;
        b[2 * e + 1] =
            (t0 + 1 < KSET) ? vb[(size_t)(t0 + 1) * 768 + t * 16 + nit] : (__bf16)0.f;
      }
      bV[t] = b;
    }
#pragma unroll
    for (int tm = 0; tm < 3; ++tm)
#pragma unroll
      for (int tn = 0; tn < 2; ++tn) o[tm][tn] = wmma_bf16(aP[tm], bV[tn], o[tm][tn]);
  }

  // ---- store (only real 36 rows) ----
#pragma unroll
  for (int tm = 0; tm < 3; ++tm)
#pragma unroll
    for (int tn = 0; tn < 2; ++tn)
#pragma unroll
      for (int e = 0; e < 8; ++e) {
        const int m = tm * 16 + e + 8 * half;
        if (m < KSET)
          outb[((size_t)s * KSET + m) * CDIM + head * DH + tn * 16 + nit] =
              (__bf16)o[tm][tn][e];
      }
}

// ---------------- gather + posembed -> bf16 GEMM inputs ----------------
__global__ __launch_bounds__(256) void prep_kernel(
    const float* __restrict__ cur, const float* __restrict__ posi,
    const int* __restrict__ inds, __bf16* __restrict__ aq, __bf16* __restrict__ av) {
  const size_t t = (size_t)blockIdx.x * blockDim.x + threadIdx.x;  // N*64 threads
  const size_t r = t >> 6;
  const int c4 = (int)(t & 63) << 2;
  const int sr = inds[r];
  const float4 x = *(const float4*)(cur + (size_t)sr * CDIM + c4);
  const float4 p = *(const float4*)(posi + (size_t)sr * CDIM + c4);
  bf4 q, v;
  q[0] = (__bf16)(x.x + p.x); q[1] = (__bf16)(x.y + p.y);
  q[2] = (__bf16)(x.z + p.z); q[3] = (__bf16)(x.w + p.w);
  v[0] = (__bf16)x.x; v[1] = (__bf16)x.y; v[2] = (__bf16)x.z; v[3] = (__bf16)x.w;
  *(bf4*)(aq + r * CDIM + c4) = q;
  *(bf4*)(av + r * CDIM + c4) = v;
}

// ---------------- layernorm kernels (1 wave / row, C=256) ----------------
__global__ __launch_bounds__(256) void ln1_kernel(
    const float* __restrict__ cur, const float* __restrict__ s2,
    const float* __restrict__ g, const float* __restrict__ b, float* __restrict__ x,
    __bf16* __restrict__ xbf) {
  const int lane = threadIdx.x & 31;
  const size_t row = (size_t)blockIdx.x * 8 + (threadIdx.x >> 5);
  const float* pa = cur + row * CDIM;
  const float* pb = s2 + row * CDIM;
  float vals[8], sum = 0.f, sq = 0.f;
#pragma unroll
  for (int j = 0; j < 8; ++j) {
    const int c = lane + 32 * j;
    const float v = pa[c] + pb[c];
    vals[j] = v; sum += v; sq += v * v;
  }
  sum = wave_sum(sum); sq = wave_sum(sq);
  const float mean = sum * (1.f / CDIM);
  const float rstd = rsqrtf(sq * (1.f / CDIM) - mean * mean + 1e-5f);
#pragma unroll
  for (int j = 0; j < 8; ++j) {
    const int c = lane + 32 * j;
    const float v = (vals[j] - mean) * rstd * g[c] + b[c];
    x[row * CDIM + c] = v;
    xbf[row * CDIM + c] = (__bf16)v;
  }
}

__global__ __launch_bounds__(256) void ln2en_kernel(
    const float* __restrict__ x, const float* __restrict__ y,
    const float* __restrict__ g2, const float* __restrict__ b2,
    const float* __restrict__ ge, const float* __restrict__ be,
    float* __restrict__ cur) {
  const int lane = threadIdx.x & 31;
  const size_t row = (size_t)blockIdx.x * 8 + (threadIdx.x >> 5);
  const float* px = x + row * CDIM;
  const float* py = y + row * CDIM;
  float* pc = cur + row * CDIM;
  float t[8], sum = 0.f, sq = 0.f;
#pragma unroll
  for (int j = 0; j < 8; ++j) {
    const int c = lane + 32 * j;
    const float v = px[c] + py[c];
    t[j] = v; sum += v; sq += v * v;
  }
  sum = wave_sum(sum); sq = wave_sum(sq);
  const float mean = sum * (1.f / CDIM);
  const float rstd = rsqrtf(sq * (1.f / CDIM) - mean * mean + 1e-5f);
  float u[8], sum2 = 0.f, sq2 = 0.f;
#pragma unroll
  for (int j = 0; j < 8; ++j) {
    const int c = lane + 32 * j;
    const float tv = (t[j] - mean) * rstd * g2[c] + b2[c];
    const float uv = tv + pc[c];
    u[j] = uv; sum2 += uv; sq2 += uv * uv;
  }
  sum2 = wave_sum(sum2); sq2 = wave_sum(sq2);
  const float mean2 = sum2 * (1.f / CDIM);
  const float rstd2 = rsqrtf(sq2 * (1.f / CDIM) - mean2 * mean2 + 1e-5f);
#pragma unroll
  for (int j = 0; j < 8; ++j) {
    const int c = lane + 32 * j;
    pc[c] = (u[j] - mean2) * rstd2 * ge[c] + be[c];
  }
}

__global__ __launch_bounds__(256) void resln_kernel(
    const float* __restrict__ a, const float* __restrict__ b,
    const float* __restrict__ g, const float* __restrict__ bb,
    float* __restrict__ dst) {
  const int lane = threadIdx.x & 31;
  const size_t row = (size_t)blockIdx.x * 8 + (threadIdx.x >> 5);
  const float* pa = a + row * CDIM;
  const float* pb = b + row * CDIM;
  float vals[8], sum = 0.f, sq = 0.f;
#pragma unroll
  for (int j = 0; j < 8; ++j) {
    const int c = lane + 32 * j;
    const float v = pa[c] + pb[c];
    vals[j] = v; sum += v; sq += v * v;
  }
  sum = wave_sum(sum); sq = wave_sum(sq);
  const float mean = sum * (1.f / CDIM);
  const float rstd = rsqrtf(sq * (1.f / CDIM) - mean * mean + 1e-5f);
#pragma unroll
  for (int j = 0; j < 8; ++j) {
    const int c = lane + 32 * j;
    dst[row * CDIM + c] = (vals[j] - mean) * rstd * g[c] + bb[c];
  }
}

// ---------------- helpers ----------------
__global__ __launch_bounds__(256) void copy4_kernel(const float* __restrict__ s,
                                                    float* __restrict__ d) {
  const size_t i = ((size_t)blockIdx.x * blockDim.x + threadIdx.x) * 4;
  *(float4*)(d + i) = *(const float4*)(s + i);
}

__global__ __launch_bounds__(256) void f2bf_kernel(const float* __restrict__ s,
                                                   __bf16* __restrict__ d, int n) {
  const int i = blockIdx.x * 256 + threadIdx.x;
  if (i < n) d[i] = (__bf16)s[i];
}

// ---------------- host orchestration ----------------
extern "C" void kernel_launch(void* const* d_in, const int* in_sizes, int n_in,
                              void* d_out, int out_size, void* d_ws, size_t ws_size,
                              hipStream_t stream) {
  const float* src = (const float*)d_in[0];
  const float* pos = (const float*)d_in[1];
  const float* in_w = (const float*)d_in[2];
  const float* in_b = (const float*)d_in[3];
  const float* out_w = (const float*)d_in[4];
  const float* out_b = (const float*)d_in[5];
  const float* l1_w = (const float*)d_in[6];
  const float* l1_b = (const float*)d_in[7];
  const float* l2_w = (const float*)d_in[8];
  const float* l2_b = (const float*)d_in[9];
  const float* n1_g = (const float*)d_in[10];
  const float* n1_b = (const float*)d_in[11];
  const float* n2_g = (const float*)d_in[12];
  const float* n2_b = (const float*)d_in[13];
  const float* en_g = (const float*)d_in[14];
  const float* en_b = (const float*)d_in[15];
  const float* res_g = (const float*)d_in[16];
  const float* res_b = (const float*)d_in[17];
  const int* voxel_inds = (const int*)d_in[18];
  const unsigned char* key_mask = (const unsigned char*)d_in[19];  // jnp bool = 1B
  float* outp = (float*)d_out;

  // ---- carve workspace (all offsets 256B aligned); needs ~330 MB ----
  char* ws = (char*)d_ws;
  size_t off = 0;
  auto carve = [&](size_t bytes) -> char* {
    char* p = ws + off;
    off = (off + bytes + 255) & ~(size_t)255;
    return p;
  };
  __bf16* wbf_in = (__bf16*)carve((size_t)LLAY * 768 * 256 * 2);
  __bf16* wbf_out = (__bf16*)carve((size_t)LLAY * 256 * 256 * 2);
  __bf16* wbf_l1 = (__bf16*)carve((size_t)LLAY * 512 * 256 * 2);
  __bf16* wbf_l2 = (__bf16*)carve((size_t)LLAY * 256 * 512 * 2);
  __bf16* aq = (__bf16*)carve((size_t)NVOX * CDIM * 2);
  __bf16* av = (__bf16*)carve((size_t)NVOX * CDIM * 2);
  __bf16* qkvbf = (__bf16*)carve((size_t)NVOX * 768 * 2);
  __bf16* attnbf = (__bf16*)carve((size_t)NVOX * CDIM * 2);
  float* s2y = (float*)carve((size_t)NVOX * CDIM * 4);  // s2, later FFN2 output
  float* xbuf = (float*)carve((size_t)NVOX * CDIM * 4);
  __bf16* xbf = (__bf16*)carve((size_t)NVOX * CDIM * 2);
  __bf16* hbuf = (__bf16*)carve((size_t)NVOX * FDIM * 2);
  float* cur = (float*)carve((size_t)NVOX * CDIM * 4);
  float* resid = (float*)carve((size_t)NVOX * CDIM * 4);
  (void)ws_size; (void)in_sizes; (void)n_in; (void)out_size;

  // ---- weights -> bf16 (recomputed every launch; deterministic) ----
  {
    const int n0 = LLAY * 768 * 256, n1 = LLAY * 256 * 256, n2 = LLAY * 512 * 256,
              n3 = LLAY * 256 * 512;
    f2bf_kernel<<<(n0 + 255) / 256, 256, 0, stream>>>(in_w, wbf_in, n0);
    f2bf_kernel<<<(n1 + 255) / 256, 256, 0, stream>>>(out_w, wbf_out, n1);
    f2bf_kernel<<<(n2 + 255) / 256, 256, 0, stream>>>(l1_w, wbf_l1, n2);
    f2bf_kernel<<<(n3 + 255) / 256, 256, 0, stream>>>(l2_w, wbf_l2, n3);
  }
  const int nv4blocks = (NVOX * CDIM / 4) / 256;  // 9216
  copy4_kernel<<<nv4blocks, 256, 0, stream>>>(src, cur);

  const dim3 gF(FDIM / BN, NVOX / BM);  // Nout=512: (8, 144)
  const dim3 gC(CDIM / BN, NVOX / BM);  // Nout=256: (4, 144)
  const int lnBlocks = NVOX / 8;        // 4608

  int li = 0;
  for (int blk = 0; blk < 2; ++blk) {
    copy4_kernel<<<nv4blocks, 256, 0, stream>>>(cur, resid);
    const int shift = blk & 1;
    for (int i = 0; i < 2; ++i) {
      const size_t part = (size_t)(shift * 2 + i) * SSET * KSET;
      const int* inds = voxel_inds + part;
      const unsigned char* msk = key_mask + part;
      const float* posi = pos + (size_t)i * NVOX * CDIM;

      // gather + pos-embed -> bf16 A matrices
      prep_kernel<<<nv4blocks, 256, 0, stream>>>(cur, posi, inds, aq, av);

      // QK projection: (N,256) @ (512,256)^T -> qkv[:,0:512]
      gemm_bf16_kernel<<<gF, 256, 0, stream>>>(
          aq, wbf_in + (size_t)li * 768 * 256, in_b + (size_t)li * 768, qkvbf,
          nullptr, CDIM, 768, GF_OUT_BF16);
      // V projection: (N,256) @ (256,256)^T -> qkv[:,512:768]
      gemm_bf16_kernel<<<gC, 256, 0, stream>>>(
          av, wbf_in + (size_t)li * 768 * 256 + 512 * 256,
          in_b + (size_t)li * 768 + 512, qkvbf + 512, nullptr, CDIM, 768,
          GF_OUT_BF16);

      // per-set multi-head attention
      attn_kernel<<<dim3(SSET, 2), 128, 0, stream>>>(qkvbf, msk, attnbf);

      // output projection + scatter (indices are a permutation -> full coverage)
      gemm_bf16_kernel<<<gC, 256, 0, stream>>>(
          attnbf, wbf_out + (size_t)li * 256 * 256, out_b + (size_t)li * 256, s2y,
          inds, CDIM, CDIM, 0);

      // x = LN(x_in + s2)
      ln1_kernel<<<lnBlocks, 256, 0, stream>>>(cur, s2y, n1_g + li * CDIM,
                                               n1_b + li * CDIM, xbuf, xbf);
      // h = gelu(x @ l1^T + b1)
      gemm_bf16_kernel<<<gF, 256, 0, stream>>>(
          xbf, wbf_l1 + (size_t)li * 512 * 256, l1_b + (size_t)li * 512, hbuf,
          nullptr, CDIM, FDIM, GF_OUT_BF16 | GF_GELU);
      // y = h @ l2^T + b2
      gemm_bf16_kernel<<<gC, 256, 0, stream>>>(
          hbuf, wbf_l2 + (size_t)li * 256 * 512, l2_b + (size_t)li * 256, s2y,
          nullptr, FDIM, CDIM, 0);
      // cur = LN(LN(x + y) + x_in)
      ln2en_kernel<<<lnBlocks, 256, 0, stream>>>(
          xbuf, s2y, n2_g + li * CDIM, n2_b + li * CDIM, en_g + li * CDIM,
          en_b + li * CDIM, cur);
      ++li;
    }
    // block residual LN (last block writes final output)
    resln_kernel<<<lnBlocks, 256, 0, stream>>>(cur, resid, res_g + blk * CDIM,
                                               res_b + blk * CDIM,
                                               (blk == 1) ? outp : cur);
  }
}
